// GridNetBlock_hs4_49005576847854
// MI455X (gfx1250) — compile-verified, matrix operand order checked by reading, not run on hardware
//
// GridNet block for MI455X (gfx1250): all GEMM-shaped work lowered to
// v_wmma_f32_16x16x32_bf16. fp32 activations are converted to bf16 for the
// matrix units; accumulation is fp32. Sequential LSTM chains are kept as
// wide single-GEMM steps to minimize the serial critical path.
// GEMM waves compute 16x64 strips (4 accumulators) so each A fragment is
// reused 4x -> ~4 wmma per 10 global_load_b128 instead of 1 per 8.
// Workspace requirement: ~570 MB in d_ws (laid out linearly below).
#include <hip/hip_runtime.h>
#include <math.h>

typedef __attribute__((ext_vector_type(16))) __bf16 v16bf;
typedef __attribute__((ext_vector_type(8)))  float  v8f;

namespace {
constexpr int B_  = 2;     // batch
constexpr int C_  = 64;    // EMB_DIM
constexpr int T_  = 1000;  // padded time (== old_T here)
constexpr int Q_  = 65;    // old_Q
constexpr int QP_ = 68;    // padded freq
constexpr int LI_ = 17;    // intra unfold length
constexpr int NI_ = 2000;  // B*T intra sequences
constexpr int MI_ = 34000; // NI*LI  (== NJ*LJ as well)
constexpr int HH_ = 128;   // intra hidden per direction
constexpr int GI_ = 512;   // 4*HH
constexpr int LJ_ = 250;   // inter unfold length
constexpr int NJ_ = 136;   // B*QP inter sequences
constexpr int NJP_= 144;   // padded to multiple of 16 for WMMA M
constexpr int HJ_ = 256;   // inter hidden (HID)
constexpr int GJ_ = 1024;  // 4*HID
constexpr int NH_ = 4, E_ = 4, D_ = 16, F_ = 65, HB_ = 8; // heads etc.
constexpr int TP_ = 1024;  // padded attention time
constexpr int KQ_ = 288;   // padded qk feature dim (260 -> 288)
constexpr int DV_ = 1040;  // v feature dim (16*65)
constexpr int MP_ = 130000;// B*T*F positions
constexpr float EPS_ = 1e-5f;
}

__device__ __forceinline__ float sigf(float x) { return 1.f / (1.f + __expf(-x)); }

__device__ __forceinline__ float blkreduce_sum(float v, float* red) {
  int tid = threadIdx.x;
  red[tid] = v; __syncthreads();
  for (int s = 128; s > 0; s >>= 1) { if (tid < s) red[tid] += red[tid + s]; __syncthreads(); }
  float r = red[0]; __syncthreads();
  return r;
}
__device__ __forceinline__ float blkreduce_max(float v, float* red) {
  int tid = threadIdx.x;
  red[tid] = v; __syncthreads();
  for (int s = 128; s > 0; s >>= 1) { if (tid < s) red[tid] = fmaxf(red[tid], red[tid + s]); __syncthreads(); }
  float r = red[0]; __syncthreads();
  return r;
}

// ---------------------------------------------------------------------------
// WMMA GEMM: C[M,N] = A[M,K] * W[N,K]^T (+bias[n]).  A,W bf16 row-major,
// C fp32 row-major.  M%16==0, N%16==0, K%32==0.
// One wave computes a 16x64 strip (4 accumulators) so the A fragment is
// reused across 4 wmma ops per K step.  4 waves / block.
// Fragment packing follows ISA 7.12.2 (16-bit A 16x32):
//   lanes 0-15: m=lane,  VGPR j holds K pair {(j/4)*16 + (j%4)*2, +1}
//   lanes 16-31: m=lane-16, same pattern +8.
//   W rows use the identical pattern with n in place of m.
//   C: lanes 0-15 n=lane m=r; lanes 16-31 n=lane-16 m=r+8.
// Edge strips clamp W pointers to subtile 0 (branch-free K loop) and guard
// only the stores.
// ---------------------------------------------------------------------------
__global__ void k_gemm(const __bf16* __restrict__ A, int lda, long long sA,
                       const __bf16* __restrict__ W, int ldw, long long sW,
                       const float* __restrict__ bias,
                       float* __restrict__ Cp, int ldc, long long sC,
                       int N, int K)
{
  int wave = threadIdx.x >> 5;
  int lane = threadIdx.x & 31;
  int strips = (N + 63) >> 6;
  int strip = blockIdx.x * 4 + wave;
  if (strip >= strips) return;
  int row0 = blockIdx.y << 4;
  int colbase = strip << 6;
  bool act1 = (colbase + 16) < N;
  bool act2 = (colbase + 32) < N;
  bool act3 = (colbase + 48) < N;
  const __bf16* Ab = A + (size_t)blockIdx.z * sA;
  const __bf16* Wb = W + (size_t)blockIdx.z * sW;
  float* Cb = Cp + (size_t)blockIdx.z * sC;
  int m  = lane & 15;
  int kh = (lane >> 4) << 3;                 // 0 or 8
  const __bf16* ap  = Ab + (size_t)(row0 + m) * lda;
  const __bf16* wp0 = Wb + (size_t)(colbase + m) * ldw;
  const __bf16* wp1 = act1 ? wp0 + (size_t)16 * ldw : wp0;
  const __bf16* wp2 = act2 ? wp0 + (size_t)32 * ldw : wp0;
  const __bf16* wp3 = act3 ? wp0 + (size_t)48 * ldw : wp0;
  v8f acc0 = {}, acc1 = {}, acc2 = {}, acc3 = {};
  for (int k0 = 0; k0 < K; k0 += 32) {
    __builtin_prefetch(ap + k0 + 64, 0, 0);   // -> global_prefetch_b8
    __builtin_prefetch(wp0 + k0 + 64, 0, 0);
    v16bf af = {}, wf0 = {}, wf1 = {}, wf2 = {}, wf3 = {};
#pragma unroll
    for (int j = 0; j < 8; ++j) {
      int kb = k0 + ((j >> 2) << 4) + kh + ((j & 3) << 1);
      af[2 * j]      = ap[kb];   af[2 * j + 1]  = ap[kb + 1];
      wf0[2 * j]     = wp0[kb];  wf0[2 * j + 1] = wp0[kb + 1];
      wf1[2 * j]     = wp1[kb];  wf1[2 * j + 1] = wp1[kb + 1];
      wf2[2 * j]     = wp2[kb];  wf2[2 * j + 1] = wp2[kb + 1];
      wf3[2 * j]     = wp3[kb];  wf3[2 * j + 1] = wp3[kb + 1];
    }
    acc0 = __builtin_amdgcn_wmma_f32_16x16x32_bf16(false, af, false, wf0,
                                                   (short)0, acc0, false, false);
    acc1 = __builtin_amdgcn_wmma_f32_16x16x32_bf16(false, af, false, wf1,
                                                   (short)0, acc1, false, false);
    acc2 = __builtin_amdgcn_wmma_f32_16x16x32_bf16(false, af, false, wf2,
                                                   (short)0, acc2, false, false);
    acc3 = __builtin_amdgcn_wmma_f32_16x16x32_bf16(false, af, false, wf3,
                                                   (short)0, acc3, false, false);
  }
  int mrow = row0 + ((lane >> 4) << 3);
  int nn = lane & 15;
  v8f accs[4] = {acc0, acc1, acc2, acc3};
#pragma unroll
  for (int i = 0; i < 4; ++i) {
    int col = colbase + 16 * i + nn;
    if (col < N) {
      float bv = bias ? bias[col] : 0.f;
#pragma unroll
      for (int r = 0; r < 8; ++r)
        Cb[(size_t)(mrow + r) * ldc + col] = accs[i][r] + bv;
    }
  }
}

// --------------------------- small utility kernels --------------------------
__global__ void k_f2b(const float* __restrict__ s, __bf16* __restrict__ d, long long n) {
  long long i = (long long)blockIdx.x * blockDim.x + threadIdx.x;
  if (i < n) d[i] = (__bf16)s[i];
}
__global__ void k_zero16(unsigned short* __restrict__ p, long long n) {
  long long i = (long long)blockIdx.x * blockDim.x + threadIdx.x;
  if (i < n) p[i] = 0;
}
__global__ void k_zerof(float* __restrict__ p, long long n) {
  long long i = (long long)blockIdx.x * blockDim.x + threadIdx.x;
  if (i < n) p[i] = 0.f;
}

// LayerNorm over channel dim of (B,C,T,q) tensor; input has inQ freq cols,
// zero-padded to QP_ cols before LN (pad cols output beta).  gamma/beta are per-channel.
__global__ void k_ln_c(const float* __restrict__ in, const float* __restrict__ gam,
                       const float* __restrict__ bet, float* __restrict__ out, int inQ)
{
  int idx = blockIdx.x * blockDim.x + threadIdx.x;
  if (idx >= B_ * T_ * QP_) return;
  int b = idx / (T_ * QP_);
  int r = idx % (T_ * QP_);
  int t = r / QP_, q = r % QP_;
  float vals[C_];
  float mu = 0.f;
#pragma unroll
  for (int c = 0; c < C_; ++c) {
    float v = (q < inQ) ? in[(((size_t)b * C_ + c) * T_ + t) * inQ + q] : 0.f;
    vals[c] = v; mu += v;
  }
  mu *= (1.f / C_);
  float var = 0.f;
#pragma unroll
  for (int c = 0; c < C_; ++c) { float d = vals[c] - mu; var += d * d; }
  var *= (1.f / C_);
  float rs = rsqrtf(var + EPS_);
#pragma unroll
  for (int c = 0; c < C_; ++c)
    out[(((size_t)b * C_ + c) * T_ + t) * QP_ + q] = (vals[c] - mu) * rs * gam[c] + bet[c];
}

// unfold intra: u[(b*T+t)*17 + l, c*4+k] = z[b,c,t,4l+k]   (bf16)
__global__ void k_pack_u_intra(const float* __restrict__ z, __bf16* __restrict__ u) {
  int idx = blockIdx.x * blockDim.x + threadIdx.x;
  if (idx >= MI_ * 256) return;
  int row = idx >> 8, col = idx & 255;
  int n = row / LI_, l = row % LI_;
  int c = col >> 2, k = col & 3;
  int b = n / T_, t = n % T_;
  int q = (l << 2) + k;
  u[idx] = (__bf16)z[(((size_t)b * C_ + c) * T_ + t) * QP_ + q];
}

// unfold inter (left-pad 3): u[(b*68+q)*250 + l, c*4+k] = z[b,c,4l+k-3,q]
__global__ void k_pack_u_inter(const float* __restrict__ z, __bf16* __restrict__ u) {
  int idx = blockIdx.x * blockDim.x + threadIdx.x;
  if (idx >= MI_ * 256) return;
  int row = idx >> 8, col = idx & 255;
  int n = row / LJ_, l = row % LJ_;
  int c = col >> 2, k = col & 3;
  int b = n / QP_, q = n % QP_;
  int tt = (l << 2) + k - 3;
  u[idx] = (__bf16)((tt >= 0) ? z[(((size_t)b * C_ + c) * T_ + tt) * QP_ + q] : 0.f);
}

// LSTM pointwise: gates_pre (Nseq,L,Gstride)+goff, tmp (Nseq_pad, 4H) recurrent
__global__ void k_lstm_pw(const float* __restrict__ gates, int Gstride, int goff,
                          const float* __restrict__ tmp, int tstride,
                          const float* __restrict__ bih, const float* __restrict__ bhh,
                          float* __restrict__ cst,
                          float* __restrict__ hseq, int Hout, int hoff,
                          __bf16* __restrict__ hbf,
                          int Nseq, int L, int H, int lio)
{
  int i = blockIdx.x * blockDim.x + threadIdx.x;
  if (i >= Nseq * H) return;
  int n = i / H, j = i % H;
  size_t gr = ((size_t)n * L + lio) * Gstride + goff;
  size_t tr = (size_t)n * tstride;
  float gi = gates[gr + j]         + tmp[tr + j]         + bih[j]         + bhh[j];
  float gf = gates[gr + H + j]     + tmp[tr + H + j]     + bih[H + j]     + bhh[H + j];
  float gg = gates[gr + 2 * H + j] + tmp[tr + 2 * H + j] + bih[2 * H + j] + bhh[2 * H + j];
  float go = gates[gr + 3 * H + j] + tmp[tr + 3 * H + j] + bih[3 * H + j] + bhh[3 * H + j];
  float cn = sigf(gf) * cst[(size_t)n * H + j] + sigf(gi) * tanhf(gg);
  float hn = sigf(go) * tanhf(cn);
  cst[(size_t)n * H + j] = cn;
  hbf[(size_t)n * H + j] = (__bf16)hn;
  hseq[((size_t)n * L + lio) * Hout + hoff + j] = hn;
}

// pack convT weight (Cin=256, Cout=64, ks=4) -> W[k*64+o, c] bf16 and bias256
__global__ void k_pack_ctw(const float* __restrict__ w, __bf16* __restrict__ wd,
                           const float* __restrict__ bs, float* __restrict__ bd)
{
  int idx = blockIdx.x * blockDim.x + threadIdx.x;
  if (idx >= 256 * 256) return;
  int j = idx >> 8, c = idx & 255;
  int k = j >> 6, o = j & 63;
  wd[(size_t)j * 256 + c] = (__bf16)w[((size_t)c * C_ + o) * 4 + k];
  if (c == 0) bd[j] = bs[o];
}

// stacked QKV weight (96,64): rows 0-15 q(h,e), 16-31 k(h,e), 32-95 v(h,d)
__global__ void k_pack_qkvw(const float* __restrict__ qw, const float* __restrict__ kw,
                            const float* __restrict__ vw, __bf16* __restrict__ dst)
{
  int idx = blockIdx.x * blockDim.x + threadIdx.x;
  if (idx >= 96 * 64) return;
  int rj = idx >> 6, c = idx & 63;
  float v;
  if (rj < 16)      v = qw[rj * 64 + c];
  else if (rj < 32) v = kw[(rj - 16) * 64 + c];
  else              v = vw[(rj - 32) * 64 + c];
  dst[idx] = (__bf16)v;
}

// intra residual: intra[b,o,t,q] = Y[(b*T+t)*17 + q/4, (q%4)*64+o] + x
__global__ void k_res_intra(const float* __restrict__ Y, const float* __restrict__ x,
                            float* __restrict__ intra)
{
  int idx = blockIdx.x * blockDim.x + threadIdx.x;
  if (idx >= B_ * C_ * T_ * QP_) return;
  int b = idx / (C_ * T_ * QP_);
  int r = idx % (C_ * T_ * QP_);
  int c = r / (T_ * QP_); r %= (T_ * QP_);
  int t = r / QP_, q = r % QP_;
  int l = q >> 2, k = q & 3;
  float v = Y[(((size_t)b * T_ + t) * LI_ + l) * 256 + (k << 6) + c];
  if (q < Q_) v += x[(((size_t)b * C_ + c) * T_ + t) * Q_ + q];
  intra[(((size_t)b * C_ + c) * T_ + t) * QP_ + q] = v;
}

// inter residual: inter[b,o,t,q] = Y[(b*68+q)*250 + t/4, (t%4)*64+o] + intra
__global__ void k_res_inter(const float* __restrict__ Y, const float* __restrict__ intra,
                            float* __restrict__ inter)
{
  int idx = blockIdx.x * blockDim.x + threadIdx.x;
  if (idx >= B_ * C_ * T_ * QP_) return;
  int b = idx / (C_ * T_ * QP_);
  int r = idx % (C_ * T_ * QP_);
  int c = r / (T_ * QP_); r %= (T_ * QP_);
  int t = r / QP_, q = r % QP_;
  int l = t >> 2, k = t & 3;
  size_t o = (((size_t)b * C_ + c) * T_ + t) * QP_ + q;
  inter[o] = Y[(((size_t)b * QP_ + q) * LJ_ + l) * 256 + (k << 6) + c] + intra[o];
}

// A_qkv[(b*T+t)*65+f, c] = inter[b,c,t,f]  (bf16)
__global__ void k_pack_aqkv(const float* __restrict__ inter, __bf16* __restrict__ a) {
  int idx = blockIdx.x * blockDim.x + threadIdx.x;
  if (idx >= MP_ * 64) return;
  int row = idx >> 6, c = idx & 63;
  int b = row / (T_ * F_);
  int r2 = row % (T_ * F_);
  int t = r2 / F_, f = r2 % F_;
  a[idx] = (__bf16)inter[(((size_t)b * C_ + c) * T_ + t) * QP_ + f];
}

// per-(head,batch,t) branch: bias + PReLU + LN over (e,f); write padded bf16
__global__ void k_branch(const float* __restrict__ ZZ, int zoff, int E,
                         const float* __restrict__ bvec, const float* __restrict__ pvec,
                         const float* __restrict__ gam, const float* __restrict__ bet,
                         __bf16* __restrict__ out, float scale, int transposed,
                         int opitch, long long obstride)
{
  __shared__ float sv[DV_];
  __shared__ float red[256];
  int t = blockIdx.x, hb = blockIdx.y;
  int h = hb >> 1, b = hb & 1;
  int nel = E * F_;
  int tid = threadIdx.x;
  float ps = 0.f;
  float a = pvec[h];
  for (int i = tid; i < nel; i += 256) {
    int e = i / F_, f = i % F_;
    float v = ZZ[((size_t)(b * T_ + t) * F_ + f) * 96 + zoff + h * E + e] + bvec[h * E + e];
    v = v >= 0.f ? v : a * v;
    sv[i] = v; ps += v;
  }
  float mu = blkreduce_sum(ps, red) / nel;
  float pv = 0.f;
  for (int i = tid; i < nel; i += 256) { float d = sv[i] - mu; pv += d * d; }
  float var = blkreduce_sum(pv, red) / nel;
  float rs = rsqrtf(var + EPS_);
  for (int i = tid; i < nel; i += 256) {
    int e = i / F_, f = i % F_;
    float v = ((sv[i] - mu) * rs * gam[(h * E + e) * F_ + f] + bet[(h * E + e) * F_ + f]) * scale;
    size_t addr = (size_t)hb * obstride +
                  (transposed ? ((size_t)i * opitch + t) : ((size_t)t * opitch + i));
    out[addr] = (__bf16)v;
  }
}

// causal softmax row (one block per (n,t)); writes bf16 attn with exact zeros
// in the masked + padded region so padded WMMA products are inert.
__global__ void k_softmax(const float* __restrict__ sc, __bf16* __restrict__ att) {
  __shared__ float red[256];
  int t = blockIdx.x, n = blockIdx.y, tid = threadIdx.x;
  const float* row = sc + ((size_t)n * TP_ + t) * TP_;
  __bf16* arow = att + ((size_t)n * TP_ + t) * TP_;
  int valid = t + 1;
  float m = -1e30f;
  for (int s = tid; s < valid; s += 256) m = fmaxf(m, row[s]);
  float mx = blkreduce_max(m, red);
  float ps = 0.f;
  for (int s = tid; s < valid; s += 256) ps += __expf(row[s] - mx);
  float inv = 1.f / blkreduce_sum(ps, red);
  for (int s = tid; s < TP_; s += 256)
    arow[s] = (__bf16)((s < valid) ? __expf(row[s] - mx) * inv : 0.f);
}

// pack av (8,1024,1040) -> A2[(b*T+t)*65+f, h*16+d] bf16
__global__ void k_pack_a2(const float* __restrict__ av, __bf16* __restrict__ a2) {
  int idx = blockIdx.x * blockDim.x + threadIdx.x;
  if (idx >= MP_ * 64) return;
  int row = idx >> 6, cd = idx & 63;
  int h = cd >> 4, d = cd & 15;
  int b = row / (T_ * F_);
  int r2 = row % (T_ * F_);
  int t = r2 / F_, f = r2 % F_;
  a2[idx] = (__bf16)av[((size_t)(h * B_ + b) * TP_ + t) * DV_ + d * F_ + f];
}

// final: PReLU(scalar) + LN over (c,f) + residual with inter, write output
__global__ void k_final(const float* __restrict__ O, const float* __restrict__ pp,
                        const float* __restrict__ gam, const float* __restrict__ bet,
                        const float* __restrict__ inter, float* __restrict__ out)
{
  __shared__ float sv[C_ * F_];  // 4160 floats
  __shared__ float red[256];
  int bt = blockIdx.x;
  int b = bt / T_, t = bt % T_;
  int tid = threadIdx.x;
  float a = pp[0];
  float ps = 0.f;
  for (int i = tid; i < C_ * F_; i += 256) {
    int c = i / F_, f = i % F_;
    float v = O[((size_t)(b * T_ + t) * F_ + f) * C_ + c];
    v = v >= 0.f ? v : a * v;
    sv[i] = v; ps += v;
  }
  float mu = blkreduce_sum(ps, red) / (C_ * F_);
  float pv = 0.f;
  for (int i = tid; i < C_ * F_; i += 256) { float d = sv[i] - mu; pv += d * d; }
  float var = blkreduce_sum(pv, red) / (C_ * F_);
  float rs = rsqrtf(var + EPS_);
  for (int i = tid; i < C_ * F_; i += 256) {
    int c = i / F_, f = i % F_;
    float v = (sv[i] - mu) * rs * gam[i] + bet[i];
    out[(((size_t)b * C_ + c) * T_ + t) * F_ + f] =
        v + inter[(((size_t)b * C_ + c) * T_ + t) * QP_ + f];
  }
}

// ------------------------------ host helpers -------------------------------
static inline unsigned cdiv(long long n, int b) { return (unsigned)((n + b - 1) / b); }

static void gemm(hipStream_t st, const __bf16* A, int lda, long long sA,
                 const __bf16* W, int ldw, long long sW,
                 const float* bias, float* Cp, int ldc, long long sC,
                 int M, int N, int K, int batch)
{
  int strips = (N + 63) / 64;
  dim3 grid((strips + 3) / 4, M / 16, batch);
  k_gemm<<<grid, 128, 0, st>>>(A, lda, sA, W, ldw, sW, bias, Cp, ldc, sC, N, K);
}

extern "C" void kernel_launch(void* const* d_in, const int* in_sizes, int n_in,
                              void* d_out, int out_size, void* d_ws, size_t ws_size,
                              hipStream_t stream)
{
  const float* x     = (const float*)d_in[0];
  const float* ig    = (const float*)d_in[1];
  const float* ib    = (const float*)d_in[2];
  const float* iwih  = (const float*)d_in[3];
  const float* iwhh  = (const float*)d_in[4];
  const float* ibih  = (const float*)d_in[5];
  const float* ibhh  = (const float*)d_in[6];
  const float* ictw  = (const float*)d_in[7];
  const float* ictb  = (const float*)d_in[8];
  const float* jg    = (const float*)d_in[9];
  const float* jb    = (const float*)d_in[10];
  const float* jwih  = (const float*)d_in[11];
  const float* jwhh  = (const float*)d_in[12];
  const float* jbih  = (const float*)d_in[13];
  const float* jbhh  = (const float*)d_in[14];
  const float* jctw  = (const float*)d_in[15];
  const float* jctb  = (const float*)d_in[16];
  const float* qw = (const float*)d_in[17];
  const float* qb = (const float*)d_in[18];
  const float* qp = (const float*)d_in[19];
  const float* qg = (const float*)d_in[20];
  const float* qbt = (const float*)d_in[21];
  const float* kw = (const float*)d_in[22];
  const float* kb = (const float*)d_in[23];
  const float* kp = (const float*)d_in[24];
  const float* kg = (const float*)d_in[25];
  const float* kbt = (const float*)d_in[26];
  const float* vw = (const float*)d_in[27];
  const float* vb = (const float*)d_in[28];
  const float* vp = (const float*)d_in[29];
  const float* vg = (const float*)d_in[30];
  const float* vbt = (const float*)d_in[31];
  const float* pw = (const float*)d_in[32];
  const float* pb = (const float*)d_in[33];
  const float* ppr = (const float*)d_in[34];
  const float* pg = (const float*)d_in[35];
  const float* pbt = (const float*)d_in[36];
  (void)in_sizes; (void)n_in; (void)out_size; (void)ws_size;

  // ---- workspace layout (linear arena) ----
  unsigned char* ws = (unsigned char*)d_ws;
  size_t off = 0;
  auto alloc = [&](size_t bytes) { size_t o = off; off += (bytes + 255) & ~(size_t)255; return o; };
  const size_t TEN = (size_t)B_ * C_ * T_ * QP_;            // 8,704,000
  float*  Z     = (float*)(ws + alloc(TEN * 4));
  float*  INTRA = (float*)(ws + alloc(TEN * 4));
  float*  INTER = (float*)(ws + alloc(TEN * 4));
  __bf16* UBF   = (__bf16*)(ws + alloc((size_t)MI_ * 256 * 2));
  float*  GATES = (float*)(ws + alloc((size_t)MI_ * GJ_ * 4));
  float*  TMP   = (float*)(ws + alloc((size_t)NI_ * GI_ * 4));
  float*  HSEQ  = (float*)(ws + alloc((size_t)MI_ * 256 * 4));
  __bf16* HBF_F = (__bf16*)(ws + alloc((size_t)NI_ * HH_ * 2));
  __bf16* HBF_B = (__bf16*)(ws + alloc((size_t)NI_ * HH_ * 2));
  float*  C_F   = (float*)(ws + alloc((size_t)NI_ * HH_ * 4));
  float*  C_B   = (float*)(ws + alloc((size_t)NI_ * HH_ * 4));
  float*  Y     = (float*)(ws + alloc((size_t)MI_ * 256 * 4));
  __bf16* WIH_I = (__bf16*)(ws + alloc((size_t)1024 * 256 * 2));
  __bf16* WHH_I = (__bf16*)(ws + alloc((size_t)1024 * 128 * 2));
  __bf16* WIH_J = (__bf16*)(ws + alloc((size_t)1024 * 256 * 2));
  __bf16* WHH_J = (__bf16*)(ws + alloc((size_t)1024 * 256 * 2));
  __bf16* CTW_I = (__bf16*)(ws + alloc((size_t)256 * 256 * 2));
  __bf16* CTW_J = (__bf16*)(ws + alloc((size_t)256 * 256 * 2));
  float*  CTB_I = (float*)(ws + alloc(256 * 4));
  float*  CTB_J = (float*)(ws + alloc(256 * 4));
  __bf16* WQKV  = (__bf16*)(ws + alloc((size_t)96 * 64 * 2));
  __bf16* WPRJ  = (__bf16*)(ws + alloc((size_t)64 * 64 * 2));
  __bf16* AQKV  = (__bf16*)(ws + alloc((size_t)MP_ * 64 * 2));
  float*  ZZ    = (float*)(ws + alloc((size_t)MP_ * 96 * 4));
  __bf16* QBF   = (__bf16*)(ws + alloc((size_t)HB_ * TP_ * KQ_ * 2));
  __bf16* KBF   = (__bf16*)(ws + alloc((size_t)HB_ * TP_ * KQ_ * 2));
  __bf16* VTB   = (__bf16*)(ws + alloc((size_t)HB_ * DV_ * TP_ * 2));
  float*  SC    = (float*)(ws + alloc((size_t)HB_ * TP_ * TP_ * 4));
  __bf16* ATT   = (__bf16*)(ws + alloc((size_t)HB_ * TP_ * TP_ * 2));
  float*  AV    = (float*)(ws + alloc((size_t)HB_ * TP_ * DV_ * 4));
  __bf16* A2    = (__bf16*)(ws + alloc((size_t)MP_ * 64 * 2));
  float*  OB    = (float*)(ws + alloc((size_t)MP_ * 64 * 4));

  // ---- weight conversions / packing ----
  k_f2b<<<cdiv(1024 * 256, 256), 256, 0, stream>>>(iwih, WIH_I, 1024 * 256);
  k_f2b<<<cdiv(1024 * 128, 256), 256, 0, stream>>>(iwhh, WHH_I, 1024 * 128);
  k_f2b<<<cdiv(1024 * 256, 256), 256, 0, stream>>>(jwih, WIH_J, 1024 * 256);
  k_f2b<<<cdiv(1024 * 256, 256), 256, 0, stream>>>(jwhh, WHH_J, 1024 * 256);
  k_pack_ctw<<<cdiv(256 * 256, 256), 256, 0, stream>>>(ictw, CTW_I, ictb, CTB_I);
  k_pack_ctw<<<cdiv(256 * 256, 256), 256, 0, stream>>>(jctw, CTW_J, jctb, CTB_J);
  k_pack_qkvw<<<cdiv(96 * 64, 256), 256, 0, stream>>>(qw, kw, vw, WQKV);
  k_f2b<<<cdiv(64 * 64, 256), 256, 0, stream>>>(pw, WPRJ, 64 * 64);

  // ---- intra path: LN -> unfold -> gate GEMM -> BiLSTM -> convT -> residual
  k_ln_c<<<cdiv((long long)B_ * T_ * QP_, 256), 256, 0, stream>>>(x, ig, ib, Z, Q_);
  k_pack_u_intra<<<cdiv((long long)MI_ * 256, 256), 256, 0, stream>>>(Z, UBF);
  gemm(stream, UBF, 256, 0, WIH_I, 256, 0, nullptr, GATES, GJ_, 0, MI_, GJ_, 256, 1);
  k_zero16<<<cdiv((long long)NI_ * HH_, 256), 256, 0, stream>>>((unsigned short*)HBF_F, (long long)NI_ * HH_);
  k_zero16<<<cdiv((long long)NI_ * HH_, 256), 256, 0, stream>>>((unsigned short*)HBF_B, (long long)NI_ * HH_);
  k_zerof<<<cdiv((long long)NI_ * HH_, 256), 256, 0, stream>>>(C_F, (long long)NI_ * HH_);
  k_zerof<<<cdiv((long long)NI_ * HH_, 256), 256, 0, stream>>>(C_B, (long long)NI_ * HH_);
  for (int s = 0; s < LI_; ++s) {
    gemm(stream, HBF_F, HH_, 0, WHH_I, HH_, 0, nullptr, TMP, GI_, 0, NI_, GI_, HH_, 1);
    k_lstm_pw<<<cdiv((long long)NI_ * HH_, 256), 256, 0, stream>>>(
        GATES, GJ_, 0, TMP, GI_, ibih, ibhh, C_F, HSEQ, 256, 0, HBF_F, NI_, LI_, HH_, s);
    gemm(stream, HBF_B, HH_, 0, WHH_I + (size_t)512 * HH_, HH_, 0, nullptr, TMP, GI_, 0, NI_, GI_, HH_, 1);
    k_lstm_pw<<<cdiv((long long)NI_ * HH_, 256), 256, 0, stream>>>(
        GATES, GJ_, GI_, TMP, GI_, ibih + GI_, ibhh + GI_, C_B, HSEQ, 256, HH_, HBF_B,
        NI_, LI_, HH_, LI_ - 1 - s);
  }
  k_f2b<<<cdiv((long long)MI_ * 256, 256), 256, 0, stream>>>(HSEQ, UBF, (long long)MI_ * 256);
  gemm(stream, UBF, 256, 0, CTW_I, 256, 0, CTB_I, Y, 256, 0, MI_, 256, 256, 1);
  k_res_intra<<<cdiv((long long)B_ * C_ * T_ * QP_, 256), 256, 0, stream>>>(Y, x, INTRA);

  // ---- inter path: LN -> unfold(pad 3) -> gate GEMM -> LSTM -> convT -> residual
  k_ln_c<<<cdiv((long long)B_ * T_ * QP_, 256), 256, 0, stream>>>(INTRA, jg, jb, Z, QP_);
  k_pack_u_inter<<<cdiv((long long)MI_ * 256, 256), 256, 0, stream>>>(Z, UBF);
  gemm(stream, UBF, 256, 0, WIH_J, 256, 0, nullptr, GATES, GJ_, 0, MI_, GJ_, 256, 1);
  k_zero16<<<cdiv((long long)NJP_ * HJ_, 256), 256, 0, stream>>>((unsigned short*)HBF_F, (long long)NJP_ * HJ_);
  k_zerof<<<cdiv((long long)NJP_ * HJ_, 256), 256, 0, stream>>>(C_F, (long long)NJP_ * HJ_);
  for (int s = 0; s < LJ_; ++s) {
    gemm(stream, HBF_F, HJ_, 0, WHH_J, HJ_, 0, nullptr, TMP, GJ_, 0, NJP_, GJ_, HJ_, 1);
    k_lstm_pw<<<cdiv((long long)NJ_ * HJ_, 256), 256, 0, stream>>>(
        GATES, GJ_, 0, TMP, GJ_, jbih, jbhh, C_F, HSEQ, 256, 0, HBF_F, NJ_, LJ_, HJ_, s);
  }
  k_f2b<<<cdiv((long long)MI_ * 256, 256), 256, 0, stream>>>(HSEQ, UBF, (long long)MI_ * 256);
  gemm(stream, UBF, 256, 0, CTW_J, 256, 0, CTB_J, Y, 256, 0, MI_, 256, 256, 1);
  k_res_inter<<<cdiv((long long)B_ * C_ * T_ * QP_, 256), 256, 0, stream>>>(Y, INTRA, INTER);

  // ---- attention ----
  k_pack_aqkv<<<cdiv((long long)MP_ * 64, 256), 256, 0, stream>>>(INTER, AQKV);
  gemm(stream, AQKV, 64, 0, WQKV, 64, 0, nullptr, ZZ, 96, 0, MP_, 96, 64, 1);
  k_zero16<<<cdiv((long long)HB_ * TP_ * KQ_, 256), 256, 0, stream>>>((unsigned short*)QBF, (long long)HB_ * TP_ * KQ_);
  k_zero16<<<cdiv((long long)HB_ * TP_ * KQ_, 256), 256, 0, stream>>>((unsigned short*)KBF, (long long)HB_ * TP_ * KQ_);
  k_zero16<<<cdiv((long long)HB_ * DV_ * TP_, 256), 256, 0, stream>>>((unsigned short*)VTB, (long long)HB_ * DV_ * TP_);
  float qscale = 1.0f / sqrtf((float)(E_ * F_));
  k_branch<<<dim3(T_, HB_), 256, 0, stream>>>(ZZ, 0, E_, qb, qp, qg, qbt, QBF, qscale, 0,
                                              KQ_, (long long)TP_ * KQ_);
  k_branch<<<dim3(T_, HB_), 256, 0, stream>>>(ZZ, 16, E_, kb, kp, kg, kbt, KBF, 1.0f, 0,
                                              KQ_, (long long)TP_ * KQ_);
  k_branch<<<dim3(T_, HB_), 256, 0, stream>>>(ZZ, 32, D_, vb, vp, vg, vbt, VTB, 1.0f, 1,
                                              TP_, (long long)DV_ * TP_);
  gemm(stream, QBF, KQ_, (long long)TP_ * KQ_, KBF, KQ_, (long long)TP_ * KQ_,
       nullptr, SC, TP_, (long long)TP_ * TP_, TP_, TP_, KQ_, HB_);
  k_softmax<<<dim3(T_, HB_), 256, 0, stream>>>(SC, ATT);
  gemm(stream, ATT, TP_, (long long)TP_ * TP_, VTB, TP_, (long long)DV_ * TP_,
       nullptr, AV, DV_, (long long)TP_ * DV_, TP_, DV_, TP_, HB_);
  k_pack_a2<<<cdiv((long long)MP_ * 64, 256), 256, 0, stream>>>(AV, A2);
  gemm(stream, A2, 64, 0, WPRJ, 64, 0, pb, OB, 64, 0, MP_, 64, 64, 1);
  k_final<<<dim3(B_ * T_), 256, 0, stream>>>(OB, ppr, pg, pbt, INTER, (float*)d_out);
}
